// MixtureOfExperts_57732950393238
// MI455X (gfx1250) — matrix operand order repaired and checked by reading, not compiled, not run on hardware
//
#include <hip/hip_runtime.h>
#include <hip/hip_bf16.h>

#define B_TOK 8192
#define D_IN  1024
#define H_DIM 4096
#define N_EXP 8
#define C_OUT 1024

typedef __attribute__((ext_vector_type(16))) _Float16 v16h;
typedef __attribute__((ext_vector_type(8)))  _Float16 v8h;
typedef __attribute__((ext_vector_type(8)))  float    v8f;
typedef __attribute__((ext_vector_type(4)))  unsigned int v4u;
typedef __attribute__((ext_vector_type(8)))  int v8i;
typedef __attribute__((ext_vector_type(4)))  int v4i;

// ---------------------------------------------------------------------------
// WMMA fragment loaders (wave32, 16-bit data, per CDNA5 ISA 7.12.2 layouts)
// ---------------------------------------------------------------------------
// A matrix 16x32 (MxK), row-major source with pitch (elements).
// lane<16: row=lane,     holds K = {0..7, 16..23}
// lane>=16: row=lane-16, holds K = {8..15, 24..31}
static __device__ __forceinline__ v16h frag_a(const _Float16* base, int pitch) {
  const int lane = threadIdx.x & 31;
  const int row  = lane & 15;
  const int koff = (lane >> 4) << 3;            // 0 or 8
  const _Float16* p = base + row * pitch + koff;
  v8h lo = *(const v8h*)(p);                     // K = koff .. koff+7
  v8h hi = *(const v8h*)(p + 16);                // K = koff+16 .. koff+23
  v16h r;
  #pragma unroll
  for (int i = 0; i < 8; ++i) { r[i] = lo[i]; r[i + 8] = hi[i]; }
  return r;
}

// B matrix 32x16 (KxN), source is N-major (transposed weights): element (k,n)
// at base + n*pitch + k, so each lane's 16 K-values are two contiguous runs of 8.
// lane: n = lane&15, koff = (lane>>4)*8, values K = koff..koff+7, koff+16..koff+23.
static __device__ __forceinline__ v16h frag_b(const _Float16* base, int pitch) {
  const int lane = threadIdx.x & 31;
  const int n  = lane & 15;
  const int kh = (lane >> 4) << 3;              // 0 or 8
  const _Float16* p = base + (long)n * pitch + kh;
  v8h lo = *(const v8h*)(p);
  v8h hi = *(const v8h*)(p + 16);
  v16h r;
  #pragma unroll
  for (int i = 0; i < 8; ++i) { r[i] = lo[i]; r[i + 8] = hi[i]; }
  return r;
}

// ---------------------------------------------------------------------------
// Gating: logits = x @ Wg + bg, softmax over E=8.  One wave per token.
// ---------------------------------------------------------------------------
__global__ void gating_kernel(const float* __restrict__ x,
                              const float* __restrict__ Wg,
                              const float* __restrict__ bg,
                              float* __restrict__ g) {
  const int wave  = threadIdx.x >> 5;
  const int lane  = threadIdx.x & 31;
  const int token = blockIdx.x * 8 + wave;
  const float* xr = x + (long)token * D_IN;

  float acc[N_EXP];
  #pragma unroll
  for (int e = 0; e < N_EXP; ++e) acc[e] = 0.0f;

  for (int d = lane; d < D_IN; d += 32) {
    const float xv = xr[d];
    const float* wr = Wg + d * N_EXP;
    #pragma unroll
    for (int e = 0; e < N_EXP; ++e) acc[e] += xv * wr[e];
  }
  #pragma unroll
  for (int off = 16; off > 0; off >>= 1) {
    #pragma unroll
    for (int e = 0; e < N_EXP; ++e) acc[e] += __shfl_xor(acc[e], off, 32);
  }
  if (lane == 0) {
    float mx = -3.0e38f;
    #pragma unroll
    for (int e = 0; e < N_EXP; ++e) { acc[e] += bg[e]; mx = acc[e] > mx ? acc[e] : mx; }
    float s = 0.0f;
    #pragma unroll
    for (int e = 0; e < N_EXP; ++e) { acc[e] = __expf(acc[e] - mx); s += acc[e]; }
    const float inv = 1.0f / s;
    #pragma unroll
    for (int e = 0; e < N_EXP; ++e) g[(long)token * N_EXP + e] = acc[e] * inv;
  }
}

// ---------------------------------------------------------------------------
// fp32 -> f16 elementwise convert (for x)
// ---------------------------------------------------------------------------
__global__ void cvt_f32_f16(const float* __restrict__ src,
                            _Float16* __restrict__ dst, long n) {
  const long i = (long)blockIdx.x * blockDim.x + threadIdx.x;
  if (i < n) dst[i] = (_Float16)src[i];
}

// ---------------------------------------------------------------------------
// Transpose + convert: src [e][R][Cc] fp32  ->  dst [e][Cc][R] f16
// ---------------------------------------------------------------------------
__global__ void transpose_cvt(const float* __restrict__ src,
                              _Float16* __restrict__ dst, int R, int Cc) {
  __shared__ _Float16 tile[32][33];
  const int e  = blockIdx.z;
  const int c0 = blockIdx.x * 32;
  const int r0 = blockIdx.y * 32;
  const int tx = threadIdx.x & 31;
  const int ty = threadIdx.x >> 5;        // 0..7
  const long base = (long)e * R * Cc;
  #pragma unroll
  for (int i = ty; i < 32; i += 8)
    tile[i][tx] = (_Float16)src[base + (long)(r0 + i) * Cc + c0 + tx];
  __syncthreads();
  #pragma unroll
  for (int i = ty; i < 32; i += 8)
    dst[base + (long)(c0 + i) * R + r0 + tx] = tile[tx][i];
}

// ---------------------------------------------------------------------------
// Fused MoE expert: per (expert, 32-token tile):
//   h = relu(x @ W1[e] + b1[e])  (chunks of 128 H-cols kept in LDS as f16)
//   eo = h @ W2[e] + b2[e]       (fp32 accumulators in VGPRs, WMMA f16)
// 256 threads = 8 waves; wave w owns C columns [w*128, w*128+128) for fc2
// and H columns [kh + w*16, kh + w*16 + 16) for fc1.
// x tile staged into LDS by the Tensor Data Mover (one 2-D descriptor).
// ---------------------------------------------------------------------------
__global__ __launch_bounds__(256, 1)
void moe_fused(const _Float16* __restrict__ xh,    // [B][D]   f16
               const _Float16* __restrict__ w1t,   // [E][H][D] f16 (transposed)
               const _Float16* __restrict__ w2t,   // [E][C][H] f16 (transposed)
               const float* __restrict__ b1,       // [E][H]
               const float* __restrict__ b2,       // [E][C]
               float* __restrict__ eo) {           // [B][E][C]
  extern __shared__ _Float16 smem[];
  _Float16* xs = smem;                  // [32][D_IN]  = 64 KB (LDS offset 0)
  _Float16* hs = smem + 32 * D_IN;      // [32][128]   =  8 KB

  const int e    = blockIdx.y;
  const int t0   = blockIdx.x * 32;
  const int tid  = threadIdx.x;
  const int wave = tid >> 5;
  const int lane = tid & 31;
  const int nn   = lane & 15;           // C/D fragment column within tile
  const int mb   = (lane >> 4) << 3;    // C/D fragment row base (0 or 8)
  const int wn0  = wave * 128;          // this wave's fc2 C-column window

  // --- TDM: DMA the 32 x D_IN f16 tile (64 KB) global -> LDS, wave 0 only ---
  if (wave == 0) {
    const unsigned long long ga =
        (unsigned long long)(const void*)(xh + (long)t0 * D_IN);
    // D# group 0: count=1, lds_addr=0, global_addr, type=2 (ISA 8.3)
    v4u g0 = { 1u,
               0u,
               (unsigned)(ga & 0xffffffffu),
               (unsigned)((ga >> 32) & 0x01ffffffu) | (2u << 30) };
    // D# group 1: data_size=2B, tensor 1024x32, tile 1024x32, stride0=1024 (ISA 8.4)
    v8i g1 = { (int)(1u << 16),            // workgroup_mask=0, data_size=1 (2B)
               (int)(D_IN << 16),          // tensor_dim0[15:0] = 1024
               (int)(32u << 16),           // tensor_dim0[31:16]=0, tensor_dim1=32
               (int)(D_IN << 16),          // tensor_dim1[31:16]=0, tile_dim0=1024
               32,                         // tile_dim1=32, tile_dim2=0
               D_IN,                       // tensor_dim0_stride[31:0] = 1024
               0, 0 };
    v4i gz4 = { 0, 0, 0, 0 };              // groups 2/3 unused (2-D tensor)
    v8i gz8 = { 0, 0, 0, 0, 0, 0, 0, 0 };  // extra group (6-arg toolchain form)
    __builtin_amdgcn_tensor_load_to_lds(g0, g1, gz4, gz4, gz8, 0);
  }
  __builtin_amdgcn_s_wait_tensorcnt(0);
  __syncthreads();

  v8f acc[16];                          // 2 M-tiles x 8 N-tiles of 16x16 fp32
  #pragma unroll
  for (int i = 0; i < 16; ++i) {
    #pragma unroll
    for (int r = 0; r < 8; ++r) acc[i][r] = 0.0f;
  }

  for (int kh = 0; kh < H_DIM; kh += 128) {
    // ---- fc1: hs[32][128] = relu(xs @ W1[e][:, kh:kh+128] + b1) ----
    // Each wave owns one 16-col N-tile; B fragment loaded ONCE, shared by
    // both M-tiles (two WMMAs per B load).
    {
      const int hcol0 = kh + wave * 16;
      const _Float16* bb = w1t + ((long)e * H_DIM + hcol0) * D_IN;
      if (kh + 128 < H_DIM)
        __builtin_prefetch((const void*)(bb + (long)128 * D_IN), 0, 1);
      v8f c0, c1;
      #pragma unroll
      for (int r = 0; r < 8; ++r) { c0[r] = 0.0f; c1[r] = 0.0f; }
      #pragma unroll 4
      for (int kk = 0; kk < D_IN; kk += 32) {
        v16h b  = frag_b(bb + kk, D_IN);
        v16h a0 = frag_a(xs + kk, D_IN);
        v16h a1 = frag_a(xs + 16 * D_IN + kk, D_IN);
        c0 = __builtin_amdgcn_wmma_f32_16x16x32_f16(false, a0, false, b,
                                                    (short)0, c0, false, false);
        c1 = __builtin_amdgcn_wmma_f32_16x16x32_f16(false, a1, false, b,
                                                    (short)0, c1, false, false);
      }
      const float bias = b1[e * H_DIM + hcol0 + nn];
      #pragma unroll
      for (int r = 0; r < 8; ++r) {
        float v0 = c0[r] + bias;
        float v1 = c1[r] + bias;
        v0 = v0 > 0.0f ? v0 : 0.0f;
        v1 = v1 > 0.0f ? v1 : 0.0f;
        hs[(mb + r) * 128 + wave * 16 + nn]      = (_Float16)v0;
        hs[(16 + mb + r) * 128 + wave * 16 + nn] = (_Float16)v1;
      }
    }
    __syncthreads();

    // ---- fc2: acc += hs @ W2[e][kh:kh+128, wn0:wn0+128] ----
    #pragma unroll
    for (int nt = 0; nt < 8; ++nt) {
      const int c0i = wn0 + nt * 16;
      const _Float16* bb = w2t + ((long)e * C_OUT + c0i) * H_DIM + kh;
      #pragma unroll
      for (int kk = 0; kk < 128; kk += 32) {
        v16h b = frag_b(bb + kk, H_DIM);
        #pragma unroll
        for (int mt = 0; mt < 2; ++mt) {
          v16h a = frag_a(hs + mt * 16 * 128 + kk, 128);
          acc[mt * 8 + nt] = __builtin_amdgcn_wmma_f32_16x16x32_f16(
              false, a, false, b, (short)0, acc[mt * 8 + nt], false, false);
        }
      }
    }
    __syncthreads();
  }

  // Epilogue: eo[t, e, c] = acc + b2[e, c]
  #pragma unroll
  for (int mt = 0; mt < 2; ++mt) {
    #pragma unroll
    for (int nt = 0; nt < 8; ++nt) {
      const int c    = wn0 + nt * 16 + nn;
      const float bv = b2[e * C_OUT + c];
      #pragma unroll
      for (int r = 0; r < 8; ++r) {
        const int tok = t0 + mt * 16 + mb + r;
        eo[((long)tok * N_EXP + e) * C_OUT + c] = acc[mt * 8 + nt][r] + bv;
      }
    }
  }
}

// ---------------------------------------------------------------------------
// mixture[b,c] = sum_e g[b,e] * eo[b,e,c]
// ---------------------------------------------------------------------------
__global__ void mixture_kernel(const float* __restrict__ g,
                               const float* __restrict__ eo,
                               float* __restrict__ mix) {
  const long i = (long)blockIdx.x * 256 + threadIdx.x;   // over B*C
  const int b = (int)(i >> 10);
  const int c = (int)(i & (C_OUT - 1));
  float s = 0.0f;
  #pragma unroll
  for (int e = 0; e < N_EXP; ++e)
    s += g[(long)b * N_EXP + e] * eo[((long)b * N_EXP + e) * C_OUT + c];
  mix[i] = s;
}

// ---------------------------------------------------------------------------
extern "C" void kernel_launch(void* const* d_in, const int* in_sizes, int n_in,
                              void* d_out, int out_size, void* d_ws, size_t ws_size,
                              hipStream_t stream) {
  (void)in_sizes; (void)n_in; (void)out_size; (void)ws_size;
  const float* x  = (const float*)d_in[0];   // [B, D]
  const float* W1 = (const float*)d_in[1];   // [E, D, H]
  const float* b1 = (const float*)d_in[2];   // [E, H]
  const float* W2 = (const float*)d_in[3];   // [E, H, C]
  const float* b2 = (const float*)d_in[4];   // [E, C]
  const float* Wg = (const float*)d_in[5];   // [D, E]
  const float* bg = (const float*)d_in[6];   // [E]

  // Output layout (tuple order): mixture [B,C], gating [B,E], expert_out [B,E,C]
  float* mix  = (float*)d_out;
  float* gate = mix  + (size_t)B_TOK * C_OUT;
  float* eo   = gate + (size_t)B_TOK * N_EXP;

  // Workspace: f16 copies (x) and f16 transposed weights. Total ~151 MB.
  _Float16* xh  = (_Float16*)d_ws;                           // [B][D]
  _Float16* w1t = xh  + (size_t)B_TOK * D_IN;                // [E][H][D]
  _Float16* w2t = w1t + (size_t)N_EXP * D_IN * H_DIM;        // [E][C][H]

  gating_kernel<<<B_TOK / 8, 256, 0, stream>>>(x, Wg, bg, gate);

  cvt_f32_f16<<<(int)(((long)B_TOK * D_IN) / 256), 256, 0, stream>>>(
      x, xh, (long)B_TOK * D_IN);
  transpose_cvt<<<dim3(H_DIM / 32, D_IN / 32, N_EXP), 256, 0, stream>>>(
      W1, w1t, D_IN, H_DIM);
  transpose_cvt<<<dim3(C_OUT / 32, H_DIM / 32, N_EXP), 256, 0, stream>>>(
      W2, w2t, H_DIM, C_OUT);

  const size_t lds_bytes = (size_t)(32 * D_IN + 32 * 128) * sizeof(_Float16); // 72 KB
  moe_fused<<<dim3(B_TOK / 32, N_EXP), 256, lds_bytes, stream>>>(
      xh, w1t, w2t, b1, b2, eo);

  mixture_kernel<<<(int)(((long)B_TOK * C_OUT) / 256), 256, 0, stream>>>(
      gate, eo, mix);
}
